// NSAttention_23210003267717
// MI455X (gfx1250) — matrix operand order repaired and checked by reading, not compiled, not run on hardware
//
#include <hip/hip_runtime.h>
#include <hip/hip_bf16.h>
#include <math.h>

// ---------------------------------------------------------------------------
// NSA-style attention for MI455X (gfx1250, wave32, WMMA + async LDS copies).
// ---------------------------------------------------------------------------

typedef __attribute__((ext_vector_type(16))) _Float16 v16h;
typedef __attribute__((ext_vector_type(8)))  _Float16 h8;
typedef __attribute__((ext_vector_type(8)))  float    v8f;
typedef int i32x4 __attribute__((vector_size(16)));

#define DMODEL 2560
#define NHEADS 32
#define HD     80
#define PH     96            // HD padded to multiple of 32 for WMMA K loops
#define BB     2
#define SS     1024
#define NTOK   (BB*SS)       // 2048
#define NB     32            // blocks per sequence
#define BSZ    32            // tokens per block
#define KTOP   16
#define SCALEF 0.1118033988749895f   // 80^-0.5
#define DHALF  (DMODEL/2)    // 1280

#ifndef __has_builtin
#define __has_builtin(x) 0
#endif
#if __has_builtin(__builtin_amdgcn_global_load_async_to_lds_b128) && \
    __has_builtin(__builtin_amdgcn_s_wait_asynccnt)
#define HAVE_ASYNC_LDS 1
#else
#define HAVE_ASYNC_LDS 0
#endif

__device__ __forceinline__ v8f vzero8() {
  v8f z;
#pragma unroll
  for (int i = 0; i < 8; ++i) z[i] = 0.f;
  return z;
}

__device__ __forceinline__ v8f wmma16x16x32(v16h a, v16h b, v8f c) {
  return __builtin_amdgcn_wmma_f32_16x16x32_f16(false, a, false, b, (short)0, c,
                                                false, false);
}

// 16-byte global -> LDS copy; async (ASYNCcnt) when the toolchain exposes the
// gfx1250 builtin, otherwise a plain load/store pair.
__device__ __forceinline__ void cp_async16(_Float16* lds_dst,
                                           const _Float16* gsrc) {
#if HAVE_ASYNC_LDS
  __builtin_amdgcn_global_load_async_to_lds_b128(
      (__attribute__((address_space(1))) i32x4*)(void*)gsrc,
      (__attribute__((address_space(3))) i32x4*)(void*)lds_dst, 0, 0);
#else
  *reinterpret_cast<h8*>(lds_dst) = *reinterpret_cast<const h8*>(gsrc);
#endif
}

__device__ __forceinline__ void cp_async_wait() {
#if HAVE_ASYNC_LDS
  __builtin_amdgcn_s_wait_asynccnt(0);
#endif
}

// A fragment (16x32 f16), row-major A[M,K], lda in elements.
// lanes 0-15: M=lane, elems 0-7 -> K 0..7, elems 8-15 -> K 16..23
// lanes 16-31: M=lane-16, elems 0-7 -> K 8..15, elems 8-15 -> K 24..31
__device__ __forceinline__ v16h load_a_frag(const _Float16* __restrict__ A,
                                            int lda, int m0, int k0, int lane) {
  int m = m0 + (lane & 15);
  int aoff = (lane & 16) ? 8 : 0;
  const _Float16* base = A + (size_t)m * lda + k0;
  h8 lo = *reinterpret_cast<const h8*>(base + aoff);
  h8 hi = *reinterpret_cast<const h8*>(base + 16 + aoff);
  v16h a;
#pragma unroll
  for (int i = 0; i < 8; ++i) { a[i] = lo[i]; a[8 + i] = hi[i]; }
  return a;
}

// B fragment (32x16 f16) from B^T stored row-major [N,K], ldb in elements.
// lanes 0-15: N=lane, elems 0-7 -> K 0..7, elems 8-15 -> K 16..23
// lanes 16-31: N=lane-16, elems 0-7 -> K 8..15, elems 8-15 -> K 24..31
__device__ __forceinline__ v16h load_b_frag(const _Float16* __restrict__ Bt,
                                            int ldb, int n0, int k0, int lane) {
  int n = n0 + (lane & 15);
  int koff = (lane & 16) ? 8 : 0;
  const _Float16* base = Bt + (size_t)n * ldb + k0 + koff;
  h8 lo = *reinterpret_cast<const h8*>(base);
  h8 hi = *reinterpret_cast<const h8*>(base + 16);
  v16h b;
#pragma unroll
  for (int i = 0; i < 8; ++i) { b[i] = lo[i]; b[8 + i] = hi[i]; }
  return b;
}

__device__ __forceinline__ float gelu_exact(float x) {
  return 0.5f * x * (1.f + erff(x * 0.7071067811865475f));
}

// ---------------------------------------------------------------------------
// elementwise helpers
// ---------------------------------------------------------------------------
__global__ void k_zero_f16(_Float16* __restrict__ p, long long n) {
  long long i = (long long)blockIdx.x * blockDim.x + threadIdx.x;
  if (i < n) p[i] = (_Float16)0.f;
}

__global__ void k_f32_to_f16(const float* __restrict__ in,
                             _Float16* __restrict__ out, long long n) {
  long long i = (long long)blockIdx.x * blockDim.x + threadIdx.x;
  if (i < n) out[i] = (_Float16)in[i];
}

// W [K,N] f32 -> Wt [N,K] f16
__global__ void k_w_transpose(const float* __restrict__ W,
                              _Float16* __restrict__ Wt, int K, int N) {
  long long i = (long long)blockIdx.x * blockDim.x + threadIdx.x;
  long long tot = (long long)K * N;
  if (i >= tot) return;
  int n = (int)(i / K), k = (int)(i % K);
  Wt[i] = (_Float16)W[(size_t)k * N + n];
}

// Build compression MLP input rows: [2048, 2560] f16.
__global__ void k_comp_prep(const _Float16* __restrict__ src, int ld,
                            const float* __restrict__ pos,
                            _Float16* __restrict__ out) {
  long long idx = (long long)blockIdx.x * blockDim.x + threadIdx.x;
  if (idx >= (long long)NTOK * DMODEL) return;
  int r = (int)(idx / DMODEL), c = (int)(idx % DMODEL);
  int bh = r >> 5, blk = r & 31;
  int i = c / HD, hd = c % HD;
  int s = blk * BSZ + i;
  float v = (float)src[((size_t)bh * SS + s) * ld + hd] + pos[c];
  out[idx] = (_Float16)v;
}

// ---------------------------------------------------------------------------
// shared GEMM epilogues
// ---------------------------------------------------------------------------
enum { EPI_F32 = 0, EPI_F16 = 1, EPI_GELU_F16 = 2, EPI_GELU_F32 = 3,
       EPI_QK96 = 4, EPI_V_VT = 5, EPI_VCT = 6 };

template <int MODE>
__device__ __forceinline__ void store_tile(v8f c, int m0, int n0, int lane,
                                           const float* __restrict__ bias,
                                           float* __restrict__ outf,
                                           _Float16* __restrict__ outh,
                                           _Float16* __restrict__ outh2,
                                           int ldo) {
  int n = n0 + (lane & 15);
  float bv = bias ? bias[n] : 0.f;
  int mofs = (lane & 16) ? 8 : 0;
#pragma unroll
  for (int r = 0; r < 8; ++r) {
    int m = m0 + r + mofs;
    float val = c[r] + bv;
    if constexpr (MODE == EPI_F32) {
      outf[(size_t)m * ldo + n] = val;
    } else if constexpr (MODE == EPI_F16) {
      outh[(size_t)m * ldo + n] = (_Float16)val;
    } else if constexpr (MODE == EPI_GELU_F16) {
      outh[(size_t)m * ldo + n] = (_Float16)gelu_exact(val);
    } else if constexpr (MODE == EPI_GELU_F32) {
      outf[(size_t)m * ldo + n] = gelu_exact(val);
    } else if constexpr (MODE == EPI_QK96) {
      int bb = m >> 10, s = m & (SS - 1);
      int h = n / HD, hd = n % HD;
      outh[(((size_t)bb * NHEADS + h) * SS + s) * PH + hd] = (_Float16)val;
    } else if constexpr (MODE == EPI_V_VT) {
      int bb = m >> 10, s = m & (SS - 1);
      int h = n / HD, hd = n % HD;
      size_t bh = (size_t)bb * NHEADS + h;
      outh[(bh * SS + s) * HD + hd] = (_Float16)val;
      outh2[(bh * HD + hd) * SS + s] = (_Float16)val;
    } else if constexpr (MODE == EPI_VCT) {
      int bh = m >> 5, blk = m & 31;
      outh[((size_t)bh * HD + n) * NB + blk] = (_Float16)val;
    }
  }
}

// ---------------------------------------------------------------------------
// Simple per-wave WMMA GEMM (used for small N, e.g. N=80)
// ---------------------------------------------------------------------------
template <int MODE>
__global__ void k_gemm(const _Float16* __restrict__ A, int lda,
                       const _Float16* __restrict__ Bt, int ldb,
                       const float* __restrict__ bias,
                       float* __restrict__ outf, _Float16* __restrict__ outh,
                       _Float16* __restrict__ outh2, int M, int N, int K,
                       int ldo) {
  int lane = threadIdx.x & 31;
  int wave = threadIdx.x >> 5;
  int tid = blockIdx.x * (blockDim.x >> 5) + wave;
  int tn = N >> 4;
  int total = (M >> 4) * tn;
  if (tid >= total) return;
  int m0 = (tid / tn) << 4;
  int n0 = (tid % tn) << 4;
  v8f c = vzero8();
  for (int k0 = 0; k0 < K; k0 += 32) {
    v16h a = load_a_frag(A, lda, m0, k0, lane);
    v16h b = load_b_frag(Bt, ldb, n0, k0, lane);
    c = wmma16x16x32(a, b, c);
  }
  store_tile<MODE>(c, m0, n0, lane, bias, outf, outh, outh2, ldo);
}

// ---------------------------------------------------------------------------
// LDS-staged, double-buffered WMMA GEMM.
// Block tile 128(M) x 64(N) x 32(K); 8 waves in a 4x2 grid, each wave owns a
// 32x32 sub-tile (2x2 WMMA frags). K-slices stream global->LDS with async
// copies (ASYNCcnt) while WMMAs consume the previous slice.
// Requires: M % 128 == 0, N % 64 == 0, K % 32 == 0.
// ---------------------------------------------------------------------------
template <int MODE>
__global__ __launch_bounds__(256) void k_gemm_tiled(
    const _Float16* __restrict__ A, int lda, const _Float16* __restrict__ Bt,
    int ldb, const float* __restrict__ bias, float* __restrict__ outf,
    _Float16* __restrict__ outh, _Float16* __restrict__ outh2, int M, int N,
    int K, int ldo) {
  __shared__ _Float16 As[2][128 * 32];
  __shared__ _Float16 Bs[2][64 * 32];
  int lane = threadIdx.x & 31, wave = threadIdx.x >> 5;
  int tnb = N / 64;
  int bm = (blockIdx.x / tnb) * 128;
  int bn = (blockIdx.x % tnb) * 64;
  int wm = wave >> 1, wn = wave & 1;

  // copy assignments: A tile 128x32 (8KB): 2 threads/row, 16 halves each.
  int ar = threadIdx.x >> 1;
  int ac = (threadIdx.x & 1) * 16;
  // B tile 64x32 (4KB): 4 threads/row, 8 halves each.
  int br = threadIdx.x >> 2;
  int bc = (threadIdx.x & 3) * 8;

  const _Float16* arow = A + (size_t)(bm + ar) * lda;
  const _Float16* brow = Bt + (size_t)(bn + br) * ldb;

  v8f acc[2][2];
#pragma unroll
  for (int i = 0; i < 2; ++i)
#pragma unroll
    for (int j = 0; j < 2; ++j) acc[i][j] = vzero8();

  int nk = K / 32;
  // prologue: stage slice 0
  cp_async16(&As[0][ar * 32 + ac], arow + ac);
  cp_async16(&As[0][ar * 32 + ac + 8], arow + ac + 8);
  cp_async16(&Bs[0][br * 32 + bc], brow + bc);
  cp_async_wait();
  __syncthreads();

  for (int kt = 0; kt < nk; ++kt) {
    int cur = kt & 1;
    if (kt + 1 < nk) {
      int k0 = (kt + 1) * 32;
      cp_async16(&As[cur ^ 1][ar * 32 + ac], arow + k0 + ac);
      cp_async16(&As[cur ^ 1][ar * 32 + ac + 8], arow + k0 + ac + 8);
      cp_async16(&Bs[cur ^ 1][br * 32 + bc], brow + k0 + bc);
    }
    v16h a0 = load_a_frag(&As[cur][0], 32, wm * 32, 0, lane);
    v16h a1 = load_a_frag(&As[cur][0], 32, wm * 32 + 16, 0, lane);
    v16h b0 = load_b_frag(&Bs[cur][0], 32, wn * 32, 0, lane);
    v16h b1 = load_b_frag(&Bs[cur][0], 32, wn * 32 + 16, 0, lane);
    acc[0][0] = wmma16x16x32(a0, b0, acc[0][0]);
    acc[0][1] = wmma16x16x32(a0, b1, acc[0][1]);
    acc[1][0] = wmma16x16x32(a1, b0, acc[1][0]);
    acc[1][1] = wmma16x16x32(a1, b1, acc[1][1]);
    cp_async_wait();
    __syncthreads();
  }
#pragma unroll
  for (int fi = 0; fi < 2; ++fi)
#pragma unroll
    for (int fj = 0; fj < 2; ++fj)
      store_tile<MODE>(acc[fi][fj], bm + wm * 32 + fi * 16,
                       bn + wn * 32 + fj * 16, lane, bias, outf, outh, outh2,
                       ldo);
}

// ---------------------------------------------------------------------------
// gates = sigmoid(gh @ gate_W2 + b2), gh [2048,1280] f32, W2 [1280,3]
// ---------------------------------------------------------------------------
__global__ void k_gates(const float* __restrict__ gh,
                        const float* __restrict__ W2,
                        const float* __restrict__ b2,
                        float* __restrict__ gates) {
  int lane = threadIdx.x & 31, wave = threadIdx.x >> 5;
  int t = blockIdx.x * 8 + wave;
  if (t >= NTOK) return;
  float s0 = 0.f, s1 = 0.f, s2 = 0.f;
  for (int i = lane; i < DHALF; i += 32) {
    float x = gh[(size_t)t * DHALF + i];
    s0 += x * W2[i * 3 + 0];
    s1 += x * W2[i * 3 + 1];
    s2 += x * W2[i * 3 + 2];
  }
  for (int off = 16; off > 0; off >>= 1) {
    s0 += __shfl_xor(s0, off, 32);
    s1 += __shfl_xor(s1, off, 32);
    s2 += __shfl_xor(s2, off, 32);
  }
  if (lane == 0) {
    gates[t * 3 + 0] = 1.f / (1.f + expf(-(s0 + b2[0])));
    gates[t * 3 + 1] = 1.f / (1.f + expf(-(s1 + b2[1])));
    gates[t * 3 + 2] = 1.f / (1.f + expf(-(s2 + b2[2])));
  }
}

// ---------------------------------------------------------------------------
// Attention pass 1: block-mean scores + per-row max (raw q.k, no scale).
// grid: B*H*16 blocks of 256 threads (8 waves). 64 queries per block.
// ---------------------------------------------------------------------------
__global__ void k_attn_pass1(const _Float16* __restrict__ qf,
                             const _Float16* __restrict__ kf,
                             float* __restrict__ bscores,
                             float* __restrict__ rowmax) {
  __shared__ float stile[64][32];
  int bh = blockIdx.x >> 4;
  int qbase = (blockIdx.x & 15) * 64;
  const _Float16* qbh = qf + (size_t)bh * SS * PH;
  const _Float16* kbh = kf + (size_t)bh * SS * PH;
  int lane = threadIdx.x & 31, wave = threadIdx.x >> 5;
  int mt = wave >> 1, nt = wave & 1;
  float rmax = -3.0e38f;
  for (int j = 0; j < NB; ++j) {
    v8f c = vzero8();
#pragma unroll
    for (int ks = 0; ks < 3; ++ks) {
      v16h a = load_a_frag(qbh, PH, qbase + mt * 16, ks * 32, lane);
      v16h b = load_b_frag(kbh, PH, j * 32 + nt * 16, ks * 32, lane);
      c = wmma16x16x32(a, b, c);
    }
    int n = nt * 16 + (lane & 15);
    int mofs = (lane & 16) ? 8 : 0;
#pragma unroll
    for (int r = 0; r < 8; ++r) stile[mt * 16 + r + mofs][n] = c[r];
    __syncthreads();
    if (threadIdx.x < 64) {
      int row = threadIdx.x;
      float sum = 0.f;
#pragma unroll
      for (int c2 = 0; c2 < 32; ++c2) {
        float sv = stile[row][c2];
        sum += sv;
        rmax = fmaxf(rmax, sv);
      }
      bscores[((size_t)bh * SS + qbase + row) * NB + j] = sum * (1.f / 32.f);
    }
    __syncthreads();
  }
  if (threadIdx.x < 64)
    rowmax[(size_t)bh * SS + qbase + threadIdx.x] = rmax;
}

// ---------------------------------------------------------------------------
// Attention pass 2: top-k mask, three softmax branches, gated combine.
// ---------------------------------------------------------------------------
__device__ __forceinline__ void a_frags_from_lds(const float et[64][32], int mt,
                                                 int lane, unsigned jbit,
                                                 const unsigned* selb, v16h& a,
                                                 v16h& am) {
  int row = mt * 16 + (lane & 15);
  int kbase = (lane & 16) ? 8 : 0;
  float msk = ((selb[row] >> jbit) & 1u) ? 1.f : 0.f;
#pragma unroll
  for (int e = 0; e < 16; ++e) {
    int kk = e + ((e & 8) ? 8 : 0) + kbase;
    float v = et[row][kk];
    a[e] = (_Float16)v;
    am[e] = (_Float16)(v * msk);
  }
}

__global__ void k_attn_pass2(const _Float16* __restrict__ qf,
                             const _Float16* __restrict__ kf,
                             const _Float16* __restrict__ vT,
                             const _Float16* __restrict__ kc,
                             const _Float16* __restrict__ vcT,
                             const float* __restrict__ bscores,
                             const float* __restrict__ rowmax,
                             const float* __restrict__ gates,
                             _Float16* __restrict__ combined) {
  __shared__ float etile[64][32];
  __shared__ float sumw[64], sums_[64], sumc[64], rowm[64];
  __shared__ unsigned selb[64];

  int bh = blockIdx.x >> 4;
  int qbase = (blockIdx.x & 15) * 64;
  int bidx = bh / NHEADS, h = bh % NHEADS;
  const _Float16* qbh = qf + (size_t)bh * SS * PH;
  const _Float16* kbh = kf + (size_t)bh * SS * PH;
  const _Float16* vTbh = vT + (size_t)bh * HD * SS;
  const _Float16* kcbh = kc + (size_t)bh * NB * PH;
  const _Float16* vcTbh = vcT + (size_t)bh * HD * NB;
  int lane = threadIdx.x & 31, wave = threadIdx.x >> 5;
  int mt = wave >> 1, nt = wave & 1;

  // --- selection mask (stable top-16 of 32 block scores) + row stats ---
  if (threadIdx.x < 64) {
    int row = threadIdx.x;
    const float* bs = bscores + ((size_t)bh * SS + qbase + row) * NB;
    float lbs[32];
#pragma unroll
    for (int i = 0; i < 32; ++i) lbs[i] = bs[i];
    unsigned bits = 0;
#pragma unroll
    for (int i = 0; i < 32; ++i) {
      int cnt = 0;
#pragma unroll
      for (int jj = 0; jj < 32; ++jj)
        cnt += (lbs[jj] > lbs[i]) || (lbs[jj] == lbs[i] && jj < i);
      if (cnt < KTOP) bits |= (1u << i);
    }
    selb[row] = bits;
    rowm[row] = rowmax[(size_t)bh * SS + qbase + row] * SCALEF;
  }
  __syncthreads();

  // owned output tiles of the 64x80 result: tile t -> mt = t&3, ntv = t>>2
  int owned[3];
  int nown = 0;
  for (int t = wave; t < 20; t += 8) owned[nown++] = t;
  v8f accw[3], accs[3];
#pragma unroll
  for (int i = 0; i < 3; ++i) { accw[i] = vzero8(); accs[i] = vzero8(); }

  float swl = 0.f, ssl = 0.f;
  for (int j = 0; j < NB; ++j) {
    // score tile [64 x 32] for key block j
    v8f c = vzero8();
#pragma unroll
    for (int ks = 0; ks < 3; ++ks) {
      v16h a = load_a_frag(qbh, PH, qbase + mt * 16, ks * 32, lane);
      v16h b = load_b_frag(kbh, PH, j * 32 + nt * 16, ks * 32, lane);
      c = wmma16x16x32(a, b, c);
    }
    {
      int n = nt * 16 + (lane & 15);
      int mofs = (lane & 16) ? 8 : 0;
#pragma unroll
      for (int r = 0; r < 8; ++r) etile[mt * 16 + r + mofs][n] = c[r];
    }
    __syncthreads();
    if (threadIdx.x < 64) {
      int row = threadIdx.x;
      float m = rowm[row];
      float sw = 0.f;
#pragma unroll
      for (int c2 = 0; c2 < 32; ++c2) {
        float e = expf(etile[row][c2] * SCALEF - m);
        etile[row][c2] = e;
        sw += e;
      }
      swl += sw;
      if ((selb[row] >> j) & 1u) ssl += sw;
    }
    __syncthreads();
    // P @ V for this key block (full + masked)
    for (int i = 0; i < nown; ++i) {
      int t = owned[i];
      int tmt = t & 3, ntv = t >> 2;
      v16h a, am;
      a_frags_from_lds(etile, tmt, lane, (unsigned)j, selb, a, am);
      v16h b = load_b_frag(vTbh, SS, ntv * 16, j * 32, lane);
      accw[i] = wmma16x16x32(a, b, accw[i]);
      accs[i] = wmma16x16x32(am, b, accs[i]);
    }
    __syncthreads();
  }
  if (threadIdx.x < 64) {
    sumw[threadIdx.x] = swl;
    sums_[threadIdx.x] = ssl;
  }
  __syncthreads();

  // --- compressed branch: q @ kc^T, softmax, @ vc ---
  {
    v8f c = vzero8();
#pragma unroll
    for (int ks = 0; ks < 3; ++ks) {
      v16h a = load_a_frag(qbh, PH, qbase + mt * 16, ks * 32, lane);
      v16h b = load_b_frag(kcbh, PH, nt * 16, ks * 32, lane);
      c = wmma16x16x32(a, b, c);
    }
    int n = nt * 16 + (lane & 15);
    int mofs = (lane & 16) ? 8 : 0;
#pragma unroll
    for (int r = 0; r < 8; ++r) etile[mt * 16 + r + mofs][n] = c[r];
  }
  __syncthreads();
  if (threadIdx.x < 64) {
    int row = threadIdx.x;
    float mx = -3.0e38f;
#pragma unroll
    for (int c2 = 0; c2 < 32; ++c2) mx = fmaxf(mx, etile[row][c2] * SCALEF);
    float sc = 0.f;
#pragma unroll
    for (int c2 = 0; c2 < 32; ++c2) {
      float e = expf(etile[row][c2] * SCALEF - mx);
      etile[row][c2] = e;
      sc += e;
    }
    sumc[row] = sc;
  }
  __syncthreads();

  v8f accc[3];
#pragma unroll
  for (int i = 0; i < 3; ++i) accc[i] = vzero8();
  for (int i = 0; i < nown; ++i) {
    int t = owned[i];
    int tmt = t & 3, ntv = t >> 2;
    v16h a, am;
    a_frags_from_lds(etile, tmt, lane, 0u, selb, a, am);  // mask unused here
    v16h b = load_b_frag(vcTbh, NB, ntv * 16, 0, lane);
    accc[i] = wmma16x16x32(a, b, accc[i]);
  }

  // --- gated combine -> combined f16 [2048, 2560] ---
  for (int i = 0; i < nown; ++i) {
    int t = owned[i];
    int tmt = t & 3, ntv = t >> 2;
    int n = ntv * 16 + (lane & 15);  // hd
    int mofs = (lane & 16) ? 8 : 0;
#pragma unroll
    for (int r = 0; r < 8; ++r) {
      int row = tmt * 16 + r + mofs;
      int s = qbase + row;
      int t0 = bidx * SS + s;
      float g0 = gates[t0 * 3 + 0];
      float g1 = gates[t0 * 3 + 1];
      float g2 = gates[t0 * 3 + 2];
      float ow = accw[i][r] / sumw[row];
      float os = accs[i][r] / sums_[row];
      float oc = accc[i][r] / sumc[row];
      combined[(size_t)t0 * DMODEL + h * HD + n] =
          (_Float16)(g0 * oc + g1 * os + g2 * ow);
    }
  }
}

// ---------------------------------------------------------------------------
// host-side launch
// ---------------------------------------------------------------------------
static inline int cdiv(long long a, long long b) { return (int)((a + b - 1) / b); }

extern "C" void kernel_launch(void* const* d_in, const int* in_sizes, int n_in,
                              void* d_out, int out_size, void* d_ws,
                              size_t ws_size, hipStream_t stream) {
  const float* hidden = (const float*)d_in[0];
  const float* Wq = (const float*)d_in[1];  const float* bq = (const float*)d_in[2];
  const float* Wk = (const float*)d_in[3];  const float* bk = (const float*)d_in[4];
  const float* Wv = (const float*)d_in[5];  const float* bv = (const float*)d_in[6];
  const float* Wo = (const float*)d_in[7];  const float* bo = (const float*)d_in[8];
  const float* comp_pos = (const float*)d_in[9];
  const float* cW1 = (const float*)d_in[10]; const float* cb1 = (const float*)d_in[11];
  const float* cW2 = (const float*)d_in[12]; const float* cb2 = (const float*)d_in[13];
  const float* gW1 = (const float*)d_in[14]; const float* gb1 = (const float*)d_in[15];
  const float* gW2 = (const float*)d_in[16]; const float* gb2 = (const float*)d_in[17];
  float* out = (float*)d_out;

  size_t off = 0;
  auto alloc = [&](size_t bytes) -> void* {
    off = (off + 255) & ~(size_t)255;
    void* p = (char*)d_ws + off;
    off += bytes;
    return p;
  };
  _Float16* hf16  = (_Float16*)alloc((size_t)NTOK * DMODEL * 2);
  _Float16* Wqt   = (_Float16*)alloc((size_t)DMODEL * DMODEL * 2);
  _Float16* Wkt   = (_Float16*)alloc((size_t)DMODEL * DMODEL * 2);
  _Float16* Wvt   = (_Float16*)alloc((size_t)DMODEL * DMODEL * 2);
  _Float16* Wot   = (_Float16*)alloc((size_t)DMODEL * DMODEL * 2);
  _Float16* gW1t  = (_Float16*)alloc((size_t)DHALF * DMODEL * 2);
  _Float16* cW1t  = (_Float16*)alloc((size_t)(4 * HD) * DMODEL * 2);
  _Float16* cW2t  = (_Float16*)alloc((size_t)HD * (4 * HD) * 2);
  _Float16* qf16  = (_Float16*)alloc((size_t)BB * NHEADS * SS * PH * 2);
  _Float16* kf16  = (_Float16*)alloc((size_t)BB * NHEADS * SS * PH * 2);
  _Float16* vf16  = (_Float16*)alloc((size_t)BB * NHEADS * SS * HD * 2);
  _Float16* vTf16 = (_Float16*)alloc((size_t)BB * NHEADS * HD * SS * 2);
  _Float16* Acomp = (_Float16*)alloc((size_t)NTOK * DMODEL * 2);
  _Float16* h1f16 = (_Float16*)alloc((size_t)NTOK * (4 * HD) * 2);
  _Float16* kcf16 = (_Float16*)alloc((size_t)BB * NHEADS * NB * PH * 2);
  _Float16* vcT   = (_Float16*)alloc((size_t)BB * NHEADS * HD * NB * 2);
  float*    ghf32 = (float*)alloc((size_t)NTOK * DHALF * 4);
  float*    gatesb= (float*)alloc((size_t)NTOK * 3 * 4);
  float*    bsc   = (float*)alloc((size_t)BB * NHEADS * SS * NB * 4);
  float*    rmax  = (float*)alloc((size_t)BB * NHEADS * SS * 4);
  _Float16* comb  = (_Float16*)alloc((size_t)NTOK * DMODEL * 2);
  (void)ws_size; (void)in_sizes; (void)n_in; (void)out_size;

  const int T = 256;
  // zero padded f16 buffers
  {
    long long nq = (long long)BB * NHEADS * SS * PH;
    k_zero_f16<<<cdiv(nq, T), T, 0, stream>>>(qf16, nq);
    k_zero_f16<<<cdiv(nq, T), T, 0, stream>>>(kf16, nq);
    long long nkc = (long long)BB * NHEADS * NB * PH;
    k_zero_f16<<<cdiv(nkc, T), T, 0, stream>>>(kcf16, nkc);
  }
  // convert + transpose
  {
    long long n = (long long)NTOK * DMODEL;
    k_f32_to_f16<<<cdiv(n, T), T, 0, stream>>>(hidden, hf16, n);
    long long nw = (long long)DMODEL * DMODEL;
    k_w_transpose<<<cdiv(nw, T), T, 0, stream>>>(Wq, Wqt, DMODEL, DMODEL);
    k_w_transpose<<<cdiv(nw, T), T, 0, stream>>>(Wk, Wkt, DMODEL, DMODEL);
    k_w_transpose<<<cdiv(nw, T), T, 0, stream>>>(Wv, Wvt, DMODEL, DMODEL);
    k_w_transpose<<<cdiv(nw, T), T, 0, stream>>>(Wo, Wot, DMODEL, DMODEL);
    long long ng = (long long)DMODEL * DHALF;
    k_w_transpose<<<cdiv(ng, T), T, 0, stream>>>(gW1, gW1t, DMODEL, DHALF);
    long long nc1 = (long long)DMODEL * (4 * HD);
    k_w_transpose<<<cdiv(nc1, T), T, 0, stream>>>(cW1, cW1t, DMODEL, 4 * HD);
    long long nc2 = (long long)(4 * HD) * HD;
    k_w_transpose<<<cdiv(nc2, T), T, 0, stream>>>(cW2, cW2t, 4 * HD, HD);
  }
  // QKV projections (tiled WMMA, async LDS staging)
  {
    int grid = (NTOK / 128) * (DMODEL / 64);  // 16*40 = 640
    k_gemm_tiled<EPI_QK96><<<grid, T, 0, stream>>>(hf16, DMODEL, Wqt, DMODEL,
        bq, nullptr, qf16, nullptr, NTOK, DMODEL, DMODEL, 0);
    k_gemm_tiled<EPI_QK96><<<grid, T, 0, stream>>>(hf16, DMODEL, Wkt, DMODEL,
        bk, nullptr, kf16, nullptr, NTOK, DMODEL, DMODEL, 0);
    k_gemm_tiled<EPI_V_VT><<<grid, T, 0, stream>>>(hf16, DMODEL, Wvt, DMODEL,
        bv, nullptr, vf16, vTf16, NTOK, DMODEL, DMODEL, 0);
  }
  // compression branch: kc then vc
  {
    long long n = (long long)NTOK * DMODEL;
    int g1 = (NTOK / 128) * ((4 * HD) / 64);  // 16*5 = 80
    int t2 = (NTOK / 16) * (HD / 16);
    k_comp_prep<<<cdiv(n, T), T, 0, stream>>>(kf16, PH, comp_pos, Acomp);
    k_gemm_tiled<EPI_GELU_F16><<<g1, T, 0, stream>>>(Acomp, DMODEL, cW1t,
        DMODEL, cb1, nullptr, h1f16, nullptr, NTOK, 4 * HD, DMODEL, 4 * HD);
    k_gemm<EPI_F16><<<cdiv(t2, 8), T, 0, stream>>>(h1f16, 4 * HD, cW2t,
        4 * HD, cb2, nullptr, kcf16, nullptr, NTOK, HD, 4 * HD, PH);
    k_comp_prep<<<cdiv(n, T), T, 0, stream>>>(vf16, HD, comp_pos, Acomp);
    k_gemm_tiled<EPI_GELU_F16><<<g1, T, 0, stream>>>(Acomp, DMODEL, cW1t,
        DMODEL, cb1, nullptr, h1f16, nullptr, NTOK, 4 * HD, DMODEL, 4 * HD);
    k_gemm<EPI_VCT><<<cdiv(t2, 8), T, 0, stream>>>(h1f16, 4 * HD, cW2t,
        4 * HD, cb2, nullptr, vcT, nullptr, NTOK, HD, 4 * HD, 0);
  }
  // gates
  {
    int gg = (NTOK / 128) * (DHALF / 64);  // 16*20 = 320
    k_gemm_tiled<EPI_GELU_F32><<<gg, T, 0, stream>>>(hf16, DMODEL, gW1t,
        DMODEL, gb1, ghf32, nullptr, nullptr, NTOK, DHALF, DMODEL, DHALF);
    k_gates<<<NTOK / 8, T, 0, stream>>>(ghf32, gW2, gb2, gatesb);
  }
  // attention
  {
    int grid = BB * NHEADS * (SS / 64);  // 1024
    k_attn_pass1<<<grid, T, 0, stream>>>(qf16, kf16, bsc, rmax);
    k_attn_pass2<<<grid, T, 0, stream>>>(qf16, kf16, vTf16, kcf16, vcT, bsc,
                                         rmax, gatesb, comb);
  }
  // output projection
  {
    int grid = (NTOK / 128) * (DMODEL / 64);
    k_gemm_tiled<EPI_F32><<<grid, T, 0, stream>>>(comb, DMODEL, Wot, DMODEL,
        bo, out, nullptr, nullptr, NTOK, DMODEL, DMODEL, DMODEL);
  }
}
